// STAttModel_12833362280585
// MI455X (gfx1250) — matrix operand order repaired and checked by reading, not compiled, not run on hardware
//
#include <hip/hip_runtime.h>
#include <hip/hip_bf16.h>
#include <math.h>

// ---------------------------------------------------------------------------
// CDNA5 (gfx1250, wave32) WMMA types: v_wmma_f32_16x16x32_bf16
// ---------------------------------------------------------------------------
typedef __attribute__((ext_vector_type(16))) __bf16 v16bf;
typedef __attribute__((ext_vector_type(8)))  float  v8f;

#define M_TOT   36864      // B*T*N = 4*24*384
#define D_MODEL 256
#define N_NODES 384
#define T_STEPS 24
#define N_HEADS 8
#define D_HEAD  32
#define SCALE_QK 0.17677669529663687f  // 1/sqrt(32)

__device__ __forceinline__ __bf16 f2bf(float f) {
    unsigned u = __builtin_bit_cast(unsigned, f);
    unsigned r = u + 0x7FFFu + ((u >> 16) & 1u);   // round-to-nearest-even
    unsigned short h = (unsigned short)(r >> 16);
    return __builtin_bit_cast(__bf16, h);
}

// ---------------------------------------------------------------------------
// Async global->LDS 16B copy (GLOBAL_LOAD_ASYNC_TO_LDS_B128, ASYNCcnt).
// Builtin prototype (from hipcc diagnostic): param0 = v4i in addrspace(1)
// (printed as "__device__"), param1 = v4i in addrspace(3), two imm ints.
// ---------------------------------------------------------------------------
#if __has_builtin(__builtin_amdgcn_global_load_async_to_lds_b128)
#define HAVE_ASYNC_LDS 1
#else
#define HAVE_ASYNC_LDS 0
#endif

typedef int v4i_b128 __attribute__((vector_size(16)));
typedef __attribute__((address_space(1))) v4i_b128* gptr_b128;
typedef __attribute__((address_space(3))) v4i_b128* lptr_b128;

__device__ __forceinline__ void async_copy_b128(const void* g, void* l) {
#if HAVE_ASYNC_LDS
    __builtin_amdgcn_global_load_async_to_lds_b128((gptr_b128)g, (lptr_b128)l, 0, 0);
#else
    *(uint4*)l = *(const uint4*)g;
#endif
}

__device__ __forceinline__ void async_wait() {
#if HAVE_ASYNC_LDS
#if __has_builtin(__builtin_amdgcn_s_wait_asynccnt)
    __builtin_amdgcn_s_wait_asynccnt(0);
#else
    asm volatile("s_wait_asynccnt 0" ::: "memory");
#endif
#endif
}

// ---------------------------------------------------------------------------
// fp32 -> bf16 bulk convert (one-time pass over X and STE)
// ---------------------------------------------------------------------------
__global__ __launch_bounds__(256)
void cvt_f32_to_bf16(const float* __restrict__ src, __bf16* __restrict__ dst, int n)
{
    const int i = (blockIdx.x * 256 + threadIdx.x) * 4;
    if (i < n) {
        const float4 v = *(const float4*)(src + i);
        dst[i + 0] = f2bf(v.x); dst[i + 1] = f2bf(v.y);
        dst[i + 2] = f2bf(v.z); dst[i + 3] = f2bf(v.w);
    }
}

// ---------------------------------------------------------------------------
// Tiled GEMM:  C[M,256] = act( A[M,Din](bf16) @ W[Din,256](fp32) + bias )
//              (+ Cadd fp32), output bf16 or fp32 (template).
// Block = 256 threads = 8 waves. Block tile 128x32; wave tile 16x32
// -> 2 WMMAs per K-chunk sharing one A fragment.
// A tiles staged via async global->LDS B128 copies; W tile (small, 8-wave
// reuse) converted fp32->bf16 synchronously, overlapping the async copies.
// ---------------------------------------------------------------------------
template<int ACT, bool HAS_CADD, bool OUT_BF16>
__global__ __launch_bounds__(256)
void gemm_bf16(const __bf16* __restrict__ A0, const __bf16* __restrict__ A1,
               int Din, const float* __restrict__ W, const float* __restrict__ bias,
               const float* __restrict__ Cadd, void* __restrict__ Cout)
{
    __shared__ __align__(16) __bf16 sA[128][32];   // [row][k]
    __shared__ __bf16 sWt[32][34];                 // [n][k] transposed, padded

    const int tid  = threadIdx.x;
    const int lane = tid & 31;
    const int wv   = tid >> 5;
    const int rowBase = blockIdx.x * 128;
    const int colBase = blockIdx.y * 32;
    const int wr = wv * 16;

    v8f acc0 = {}, acc1 = {};
    const int nCh = Din >> 5;

    for (int ch = 0; ch < nCh; ++ch) {
        const int kbase = ch * 32;
        const __bf16* Ap; int koff;
        if (kbase < 256) { Ap = A0; koff = kbase; }
        else             { Ap = A1; koff = kbase - 256; }

        // ---- async stage A: 128 rows x 64B = 512 B128 segs, 2 per thread --
        #pragma unroll
        for (int s = 0; s < 2; ++s) {
            const int seg  = tid + s * 256;
            const int r    = seg >> 2;
            const int part = seg & 3;
            async_copy_b128(Ap + (rowBase + r) * 256 + koff + part * 8,
                            &sA[r][part * 8]);
        }
        // ---- stage W tile transposed (32x32 fp32->bf16), overlaps async ----
        {
            const int k  = tid >> 3;
            const int n0 = (tid & 7) * 4;
            const float* src = W + (kbase + k) * 256 + colBase + n0;
            #pragma unroll
            for (int j = 0; j < 4; ++j) sWt[n0 + j][k] = f2bf(src[j]);
            if (ch + 1 < nCh)
                __builtin_prefetch(W + (kbase + 32 + k) * 256 + colBase + n0, 0, 0);
        }
        async_wait();
        __syncthreads();

        // ---- fragments (ISA 7.12.2: 16-bit A 16x32; B 32x16) ---------------
        v16bf af, b0, b1;
        const int arow = wr + (lane & 15);
        const int akb  = (lane >> 4) * 16;
        #pragma unroll
        for (int j = 0; j < 16; ++j) af[j] = sA[arow][j + akb];
        const int bc  = lane & 15;
        const int bkb = (lane >> 4) * 16;
        #pragma unroll
        for (int j = 0; j < 16; ++j) b0[j] = sWt[bc][bkb + j];
        #pragma unroll
        for (int j = 0; j < 16; ++j) b1[j] = sWt[16 + bc][bkb + j];

        acc0 = __builtin_amdgcn_wmma_f32_16x16x32_bf16(
                   false, af, false, b0, (short)0, acc0, false, false);
        acc1 = __builtin_amdgcn_wmma_f32_16x16x32_bf16(
                   false, af, false, b1, (short)0, acc1, false, false);
        __syncthreads();
    }

    // ---- branch-free epilogue (template-specialized) -----------------------
    const int mo  = (lane < 16) ? 0 : 8;
    const int nc0 = colBase + (lane & 15);
    const float bv0 = bias[nc0];
    const float bv1 = bias[nc0 + 16];
    #pragma unroll
    for (int r = 0; r < 8; ++r) {
        const int m = rowBase + wr + r + mo;
        float f0 = acc0[r] + bv0;
        float f1 = acc1[r] + bv1;
        if (HAS_CADD) {
            f0 += Cadd[m * 256 + nc0];
            f1 += Cadd[m * 256 + nc0 + 16];
        }
        if (ACT == 1) { f0 = fmaxf(f0, 0.0f); f1 = fmaxf(f1, 0.0f); }
        if (OUT_BF16) {
            __bf16* C = (__bf16*)Cout;
            C[m * 256 + nc0]      = f2bf(f0);
            C[m * 256 + nc0 + 16] = f2bf(f1);
        } else {
            float* C = (float*)Cout;
            C[m * 256 + nc0]      = f0;
            C[m * 256 + nc0 + 16] = f1;
        }
    }
}

// ---------------------------------------------------------------------------
// Spatial attention: per (head, b*t, 16-query tile) over 384 nodes.
// 64 threads = 2 waves. bf16 in/out. QK^T and PV via WMMA; fp32 softmax in LDS.
// Q and per-wave K tiles staged with async B128 copies.
// ---------------------------------------------------------------------------
__global__ __launch_bounds__(64)
void spatial_attn(const __bf16* __restrict__ Q, const __bf16* __restrict__ Kt,
                  const __bf16* __restrict__ V, __bf16* __restrict__ O)
{
    const int rt = blockIdx.x;         // query row tile (0..23)
    const int bt = blockIdx.y;         // b*T + t (0..95)
    const int h  = blockIdx.z;         // head
    const int base = bt * N_NODES;
    const int hc   = h * D_HEAD;
    const int tid  = threadIdx.x;
    const int lane = tid & 31;
    const int wv   = tid >> 5;

    __shared__ __align__(16) __bf16 sQ[16][32];       // [qrow][d]
    __shared__ __align__(16) __bf16 sK[2][16][32];    // per-wave [key][d]
    __shared__ float  sS[16][388];                    // fp32 scores (padded)
    __shared__ __bf16 sP[16][384];                    // bf16 probs
    __shared__ __bf16 sVt[32][34];                    // [d][key] current chunk
    __shared__ float  red[16][4];
    __shared__ float  rowmax[16], rowinv[16];

    // stage Q rows (16x64B) via async, 1 seg/thread
    {
        const int r    = tid >> 2;
        const int part = tid & 3;
        async_copy_b128(Q + (base + rt * 16 + r) * 256 + hc + part * 8,
                        &sQ[r][part * 8]);
    }
    async_wait();
    __syncthreads();

    const int arow = lane & 15;
    const int akb  = (lane >> 4) * 16;
    const int bkb  = (lane >> 4) * 16;

    // ---- phase 1: scores = Q K^T * scale (24 key tiles, 12 per wave) -------
    for (int jj = 0; jj < 12; ++jj) {
        const int jt = jj * 2 + wv;                  // wave-private key tile
        #pragma unroll
        for (int s = 0; s < 2; ++s) {                // 64 segs / 32 lanes
            const int seg  = lane + s * 32;
            const int r    = seg >> 2;
            const int part = seg & 3;
            async_copy_b128(Kt + (base + jt * 16 + r) * 256 + hc + part * 8,
                            &sK[wv][r][part * 8]);
        }
        async_wait();                                // wave-level: no barrier

        v16bf af, bf_;
        #pragma unroll
        for (int j = 0; j < 16; ++j) af[j] = sQ[arow][j + akb];
        #pragma unroll
        for (int j = 0; j < 16; ++j) bf_[j] = sK[wv][lane & 15][bkb + j];
        v8f sc = {};
        sc = __builtin_amdgcn_wmma_f32_16x16x32_bf16(
                 false, af, false, bf_, (short)0, sc, false, false);
        const int mo  = (lane < 16) ? 0 : 8;
        const int col = jt * 16 + (lane & 15);
        #pragma unroll
        for (int r = 0; r < 8; ++r) sS[r + mo][col] = sc[r] * SCALE_QK;
    }
    __syncthreads();

    // ---- phase 2: softmax over 384 (4 threads per row) ---------------------
    {
        const int row  = tid & 15;
        const int part = tid >> 4;
        float mx = -3.0e38f;
        for (int c = part; c < 384; c += 4) mx = fmaxf(mx, sS[row][c]);
        red[row][part] = mx;
        __syncthreads();
        if (part == 0)
            rowmax[row] = fmaxf(fmaxf(red[row][0], red[row][1]),
                                fmaxf(red[row][2], red[row][3]));
        __syncthreads();
        const float m = rowmax[row];
        float sum = 0.0f;
        for (int c = part; c < 384; c += 4) {
            const float e = __expf(sS[row][c] - m);
            sS[row][c] = e;
            sum += e;
        }
        red[row][part] = sum;
        __syncthreads();
        if (part == 0)
            rowinv[row] = 1.0f / (red[row][0] + red[row][1] +
                                  red[row][2] + red[row][3]);
        __syncthreads();
        const float inv = rowinv[row];
        for (int c = part; c < 384; c += 4) sP[row][c] = f2bf(sS[row][c] * inv);
    }

    // ---- phase 3: out = P @ V (12 chunks of 32 keys; wave owns 16 cols) ----
    v8f oacc = {};
    for (int c = 0; c < 12; ++c) {
        __syncthreads();                       // protect sVt reuse
        {
            const int key = tid >> 1;          // 32 keys x 32 dims
            const int d0  = (tid & 1) * 16;
            const __bf16* src = V + (base + c * 32 + key) * 256 + hc + d0;
            #pragma unroll
            for (int j = 0; j < 16; ++j) sVt[d0 + j][key] = src[j];
        }
        __syncthreads();
        v16bf af, bf_;
        #pragma unroll
        for (int j = 0; j < 16; ++j)
            af[j] = sP[arow][c * 32 + (j + akb)];
        const int dcol = wv * 16 + (lane & 15);
        #pragma unroll
        for (int j = 0; j < 16; ++j) bf_[j] = sVt[dcol][bkb + j];
        oacc = __builtin_amdgcn_wmma_f32_16x16x32_bf16(
                   false, af, false, bf_, (short)0, oacc, false, false);
    }
    const int mo   = (lane < 16) ? 0 : 8;
    const int ncol = hc + wv * 16 + (lane & 15);
    #pragma unroll
    for (int r = 0; r < 8; ++r)
        O[(base + rt * 16 + r + mo) * 256 + ncol] = f2bf(oacc[r]);
}

// ---------------------------------------------------------------------------
// Temporal attention: per (node, b, head) over T=24 steps (0.9 GF -> VALU).
// bf16 in/out, fp32 math. K/V staged in LDS.
// ---------------------------------------------------------------------------
__global__ __launch_bounds__(32)
void temporal_attn(const __bf16* __restrict__ Q, const __bf16* __restrict__ Kt,
                   const __bf16* __restrict__ V, __bf16* __restrict__ O)
{
    const int n = blockIdx.x, b = blockIdx.y, h = blockIdx.z;
    const int hc = h * D_HEAD;
    const int tid = threadIdx.x;

    __shared__ float sK[24][32];
    __shared__ float sV[24][32];
    for (int i = tid; i < 24 * 32; i += 32) {
        const int s = i >> 5, dd = i & 31;
        const int row = ((b * T_STEPS + s) * N_NODES + n) * 256 + hc + dd;
        sK[s][dd] = (float)Kt[row];
        sV[s][dd] = (float)V[row];
    }
    __syncthreads();

    if (tid < 24) {
        const int s = tid;
        const int qrow = ((b * T_STEPS + s) * N_NODES + n) * 256 + hc;
        float q[32];
        #pragma unroll
        for (int dd = 0; dd < 32; ++dd) q[dd] = (float)Q[qrow + dd];

        float sc[24];
        float mx = -3.0e38f;
        for (int s2 = 0; s2 < 24; ++s2) {
            float a = 0.0f;
            #pragma unroll
            for (int dd = 0; dd < 32; ++dd) a += q[dd] * sK[s2][dd];
            sc[s2] = a * SCALE_QK;
            mx = fmaxf(mx, sc[s2]);
        }
        float sum = 0.0f;
        for (int s2 = 0; s2 < 24; ++s2) {
            const float e = __expf(sc[s2] - mx);
            sc[s2] = e; sum += e;
        }
        const float inv = 1.0f / sum;
        float out[32];
        #pragma unroll
        for (int dd = 0; dd < 32; ++dd) out[dd] = 0.0f;
        for (int s2 = 0; s2 < 24; ++s2) {
            const float p = sc[s2];
            #pragma unroll
            for (int dd = 0; dd < 32; ++dd) out[dd] += p * sV[s2][dd];
        }
        #pragma unroll
        for (int dd = 0; dd < 32; ++dd) O[qrow + dd] = f2bf(out[dd] * inv);
    }
}

// ---------------------------------------------------------------------------
// Gated fusion: H = sigmoid(Z)*HS + (1-sigmoid(Z))*HT  (Z fp32; rest bf16)
// ---------------------------------------------------------------------------
__global__ __launch_bounds__(256)
void fuse_gate(const float* __restrict__ Z, const __bf16* __restrict__ HS,
               const __bf16* __restrict__ HT, __bf16* __restrict__ H, int ntot)
{
    const int i = blockIdx.x * 256 + threadIdx.x;
    if (i < ntot) {
        const float z = 1.0f / (1.0f + __expf(-Z[i]));
        H[i] = f2bf(z * (float)HS[i] + (1.0f - z) * (float)HT[i]);
    }
}

// ---------------------------------------------------------------------------
extern "C" void kernel_launch(void* const* d_in, const int* in_sizes, int n_in,
                              void* d_out, int out_size, void* d_ws, size_t ws_size,
                              hipStream_t stream)
{
    const float* X   = (const float*)d_in[0];
    const float* STE = (const float*)d_in[1];
    const float* W7  = (const float*)d_in[2];  const float* b7  = (const float*)d_in[3];
    const float* W8  = (const float*)d_in[4];  const float* b8  = (const float*)d_in[5];
    const float* W9  = (const float*)d_in[6];  const float* b9  = (const float*)d_in[7];
    const float* W10 = (const float*)d_in[8];  const float* b10 = (const float*)d_in[9];
    const float* W11 = (const float*)d_in[10]; const float* b11 = (const float*)d_in[11];
    const float* W12 = (const float*)d_in[12]; const float* b12 = (const float*)d_in[13];
    const float* W13 = (const float*)d_in[14]; const float* b13 = (const float*)d_in[15];
    const float* W14 = (const float*)d_in[16]; const float* b14 = (const float*)d_in[17];
    const float* W15 = (const float*)d_in[18]; const float* b15 = (const float*)d_in[19];
    const float* W16 = (const float*)d_in[20]; const float* b16 = (const float*)d_in[21];
    const float* W17 = (const float*)d_in[22]; const float* b17 = (const float*)d_in[23];
    const float* W18 = (const float*)d_in[24]; const float* b18 = (const float*)d_in[25];
    const float* W19 = (const float*)d_in[26]; const float* b19 = (const float*)d_in[27];
    const float* W20 = (const float*)d_in[28]; const float* b20 = (const float*)d_in[29];

    const size_t S = (size_t)M_TOT * D_MODEL;
    __bf16* bb = (__bf16*)d_ws;
    __bf16* b0 = bb + 0 * S;   // Xb
    __bf16* b1 = bb + 1 * S;   // STEb
    __bf16* b2 = bb + 2 * S;
    __bf16* b3 = bb + 3 * S;
    __bf16* b4 = bb + 4 * S;
    __bf16* b5 = bb + 5 * S;
    __bf16* b6 = bb + 6 * S;
    __bf16* b7b = bb + 7 * S;
    float*  zf = (float*)(bb + 8 * S);   // fp32 gate pre-activation

    const dim3 gG(M_TOT / 128, D_MODEL / 32);            // 288 x 8
    const dim3 bG(256);
    const dim3 gS(N_NODES / 16, 4 * T_STEPS, N_HEADS);   // 24 x 96 x 8
    const dim3 bS(64);
    const dim3 gT(N_NODES, 4, N_HEADS);                  // 384 x 4 x 8
    const dim3 bT(32);
    const dim3 gE((unsigned)((S + 255) / 256));
    const dim3 gC((unsigned)(S / 4 / 256));

    // one-time bf16 conversion of inputs
    cvt_f32_to_bf16<<<gC, bG, 0, stream>>>(X,   b0, (int)S);
    cvt_f32_to_bf16<<<gC, bG, 0, stream>>>(STE, b1, (int)S);

    // ---- spatial branch ----------------------------------------------------
    gemm_bf16<1,false,true ><<<gG, bG, 0, stream>>>(b0, b1, 512, W7,  b7,  nullptr, b2); // Qs
    gemm_bf16<1,false,true ><<<gG, bG, 0, stream>>>(b0, b1, 512, W8,  b8,  nullptr, b3); // Ks
    gemm_bf16<1,false,true ><<<gG, bG, 0, stream>>>(b0, b1, 512, W9,  b9,  nullptr, b4); // Vs
    spatial_attn<<<gS, bS, 0, stream>>>(b2, b3, b4, b5);                                 // AOs
    gemm_bf16<1,false,true ><<<gG, bG, 0, stream>>>(b5, nullptr, 256, W10, b10, nullptr, b2); // G1
    gemm_bf16<0,false,true ><<<gG, bG, 0, stream>>>(b2, nullptr, 256, W11, b11, nullptr, b3); // HS

    // ---- temporal branch ---------------------------------------------------
    gemm_bf16<1,false,true ><<<gG, bG, 0, stream>>>(b0, b1, 512, W12, b12, nullptr, b4); // Qt
    gemm_bf16<1,false,true ><<<gG, bG, 0, stream>>>(b0, b1, 512, W13, b13, nullptr, b5); // Kt
    gemm_bf16<1,false,true ><<<gG, bG, 0, stream>>>(b0, b1, 512, W14, b14, nullptr, b6); // Vt
    temporal_attn<<<gT, bT, 0, stream>>>(b4, b5, b6, b7b);                               // AOt
    gemm_bf16<1,false,true ><<<gG, bG, 0, stream>>>(b7b, nullptr, 256, W15, b15, nullptr, b4); // G2
    gemm_bf16<0,false,true ><<<gG, bG, 0, stream>>>(b4,  nullptr, 256, W16, b16, nullptr, b5); // HT

    // ---- gated fusion ------------------------------------------------------
    gemm_bf16<0,false,false><<<gG, bG, 0, stream>>>(b3, nullptr, 256, W17, b17, nullptr, zf); // Z  = HS@W17
    gemm_bf16<0,true ,false><<<gG, bG, 0, stream>>>(b5, nullptr, 256, W18, b18, zf,      zf); // Z += HT@W18
    fuse_gate<<<gE, bG, 0, stream>>>(zf, b3, b5, b6, (int)S);                                 // H

    // ---- output projection + residual --------------------------------------
    gemm_bf16<1,false,true ><<<gG, bG, 0, stream>>>(b6, nullptr, 256, W19, b19, nullptr, b2); // G3
    gemm_bf16<0,true ,false><<<gG, bG, 0, stream>>>(b2, nullptr, 256, W20, b20, X, d_out);    // X + G3@W20
}
